// NeuralMemory_14096082666175
// MI455X (gfx1250) — matrix operand order repaired and checked by reading, not compiled, not run on hardware
//
#include <hip/hip_runtime.h>
#include <hip/hip_bf16.h>
#include <math.h>

// Shapes from the reference
#define Bn  4
#define Sn  1024
#define Dn  1024
#define DKn 64
#define Hn  256
#define BSn (Bn * Sn)   // 4096

typedef __attribute__((ext_vector_type(16))) _Float16 v16h;
typedef __attribute__((ext_vector_type(8)))  float    v8f;
typedef __attribute__((ext_vector_type(4)))  unsigned int u32x4;
typedef __attribute__((ext_vector_type(8)))  int      i32x8;
typedef __attribute__((ext_vector_type(4)))  int      i32x4;

// ---------------------------------------------------------------------------
// WMMA fragment loaders (CDNA5 16-bit layouts, f32 -> f16 convert on the fly)
// A (16x32, MxK): lane L<16 holds row L, K = {k0+0..7, k0+16..23};
//                 lane L>=16 holds row L-16, K = {k0+8..15, k0+24..31}
// B (32x16, KxN): lane L<16 holds col L, K = {k0+0..7, k0+16..23};
//                 lane L>=16 holds col L-16, K = {k0+8..15, k0+24..31}
// C/D (16x16 f32): vgpr r -> M = r + (lane<16 ? 0 : 8), N = lane%16
// ---------------------------------------------------------------------------
__device__ __forceinline__ v16h load_a_frag(const float* __restrict__ base,
                                            int lda, int k0, int lane) {
  v16h a;
  int r    = lane & 15;
  int koff = (lane < 16) ? 0 : 8;
  const float* p = base + r * lda + k0 + koff;
#pragma unroll
  for (int e = 0; e < 8; ++e) a[e] = (_Float16)p[e];
#pragma unroll
  for (int e = 0; e < 8; ++e) a[8 + e] = (_Float16)p[16 + e];
  return a;
}

__device__ __forceinline__ v16h load_b_frag(const float* __restrict__ w,
                                            int ldw, int n0, int k0, int lane) {
  v16h b;
  int col = lane & 15;
  int kb  = k0 + ((lane < 16) ? 0 : 8);
  const float* p = w + (n0 + col) * ldw + kb;
#pragma unroll
  for (int e = 0; e < 8; ++e) b[e] = (_Float16)p[e];
#pragma unroll
  for (int e = 0; e < 8; ++e) b[8 + e] = (_Float16)p[16 + e];
  return b;
}

// ---------------------------------------------------------------------------
// Tensor Data Mover: DMA one 1-D tile of 192 f32 (k|v|q for one timestep)
// from global memory into LDS. D# packed per CDNA5 ISA (CDNA5 ISA).
// LDS offset = low 32 bits of the flat pointer (LDS aperture maps addr[31:0]).
// This toolchain exposes the 6-arg builtin: (g0,g1,g2,g3,g4,cpol).
// ---------------------------------------------------------------------------
__device__ __forceinline__ void tdm_load_kvq(const float* gsrc, float* ldst) {
  unsigned long long ga  = (unsigned long long)(const void*)gsrc;
  unsigned           lds = (unsigned)(unsigned long long)(void*)ldst;
  u32x4 g0;
  g0[0] = 1u;                                        // count=1, user descriptor
  g0[1] = lds;                                       // lds_addr (bytes)
  g0[2] = (unsigned)(ga & 0xffffffffu);              // global_addr[31:0]
  g0[3] = (unsigned)((ga >> 32) & 0x01ffffffu)       // global_addr[56:32]
        | (2u << 30);                                // type = 2 ("image")
  i32x8 g1;
  g1[0] = (int)(2u << 16);                           // data_size = 4B
  g1[1] = (int)(192u << 16);                         // tensor_dim0 = 192 (lo16 @63:48)
  g1[2] = (int)(1u << 16);                           // tensor_dim0 hi=0 | tensor_dim1 = 1
  g1[3] = (int)(192u << 16);                         // tensor_dim1 hi=0 | tile_dim0 = 192
  g1[4] = 1;                                         // tile_dim1 = 1, tile_dim2 = 0
  g1[5] = 192;                                       // tensor_dim0_stride lo = 192
  g1[6] = 0;
  g1[7] = 0;                                         // tensor_dim1_stride = 0
  i32x4 g2; g2[0] = 1; g2[1] = 1; g2[2] = 0; g2[3] = 0;   // dim2=dim3=1
  i32x4 g3; g3[0] = 0; g3[1] = 0; g3[2] = 0; g3[3] = 0;
  i32x8 g4; g4[0] = 0; g4[1] = 0; g4[2] = 0; g4[3] = 0;
            g4[4] = 0; g4[5] = 0; g4[6] = 0; g4[7] = 0;
  __builtin_amdgcn_tensor_load_to_lds(g0, g1, g2, g3, g4, 0);
}

// ---------------------------------------------------------------------------
// Kernel 1: P[bs, 0:192] = x @ [Wk;Wv;Wq]^T + bias   (WMMA, f32 acc)
// grid 768 x block 128 (4 waves); tile = 16x16, K-loop 1024/32
// ---------------------------------------------------------------------------
__global__ void __launch_bounds__(128)
proj_kernel(const float* __restrict__ x,
            const float* __restrict__ Wk, const float* __restrict__ bk,
            const float* __restrict__ Wv, const float* __restrict__ bv,
            const float* __restrict__ Wq, const float* __restrict__ bq,
            float* __restrict__ P) {
  int wave = threadIdx.x >> 5;
  int lane = threadIdx.x & 31;
  int tile = blockIdx.x * 4 + wave;            // 256 row-tiles * 12 col-tiles
  if (tile >= (BSn / 16) * 12) return;
  int rowt = tile / 12, colt = tile % 12;
  int row0 = rowt * 16, n0 = colt * 16;

  const float* w; const float* bias; int nloc;
  if (n0 < 64)        { w = Wk; bias = bk; nloc = n0;       }
  else if (n0 < 128)  { w = Wv; bias = bv; nloc = n0 - 64;  }
  else                { w = Wq; bias = bq; nloc = n0 - 128; }

  v8f c = {};
#pragma unroll 2
  for (int k0 = 0; k0 < Dn; k0 += 32) {
    v16h a = load_a_frag(x + row0 * Dn, Dn, k0, lane);
    v16h b = load_b_frag(w, Dn, nloc, k0, lane);
    c = __builtin_amdgcn_wmma_f32_16x16x32_f16(false, a, false, b,
                                               (short)0, c, false, false);
  }
  int   col   = lane & 15;
  float badd  = bias[nloc + col];
  int   mbase = (lane < 16) ? 0 : 8;
#pragma unroll
  for (int r = 0; r < 8; ++r)
    P[(row0 + mbase + r) * 192 + n0 + col] = c[r] + badd;
}

// ---------------------------------------------------------------------------
// Kernel 2: gates G[bs, 0:6] = sigmoid(x @ [Wlr;Wmom;Wfg]^T + b)
// grid 4096 x block 192 (6 waves; one wave per gate column)
// ---------------------------------------------------------------------------
__global__ void __launch_bounds__(192)
gates_kernel(const float* __restrict__ x,
             const float* __restrict__ Wlr,  const float* __restrict__ blr,
             const float* __restrict__ Wmom, const float* __restrict__ bmom,
             const float* __restrict__ Wfg,  const float* __restrict__ bfg,
             float* __restrict__ G) {
  int bs   = blockIdx.x;
  int j    = threadIdx.x >> 5;
  int lane = threadIdx.x & 31;
  const float* w; float bb;
  if (j < 2)      { w = Wlr  + j * Dn;       bb = blr[j];      }
  else if (j < 4) { w = Wmom + (j - 2) * Dn; bb = bmom[j - 2]; }
  else            { w = Wfg  + (j - 4) * Dn; bb = bfg[j - 4];  }
  const float* xr = x + bs * Dn;
  float p = 0.f;
  for (int i = lane; i < Dn; i += 32) p += xr[i] * w[i];
#pragma unroll
  for (int off = 16; off >= 1; off >>= 1) p += __shfl_xor(p, off);
  if (lane == 0) G[bs * 6 + j] = 1.f / (1.f + __expf(-(p + bb)));
}

// ---------------------------------------------------------------------------
// Kernel 3: causal depthwise conv (K=4) + L2-normalize(k, q)
// Writes the fused KVQ stream: KVQ[bs][0:64]=k_norm, [64:128]=v, [128:192]=q_norm
// grid 4096 (one per (b,s)) x block 64 (one thread per channel)
// ---------------------------------------------------------------------------
__global__ void __launch_bounds__(64)
conv_kernel(const float* __restrict__ P,
            const float* __restrict__ kw, const float* __restrict__ kb,
            const float* __restrict__ vw, const float* __restrict__ vb,
            const float* __restrict__ qw, const float* __restrict__ qb,
            float* __restrict__ KVQ) {
  __shared__ float redk[64];
  __shared__ float redq[64];
  int bs = blockIdx.x;
  int b  = bs >> 10, s = bs & 1023;
  int c  = threadIdx.x;
  float ka = kb[c], va = vb[c], qa = qb[c];
#pragma unroll
  for (int j = 0; j < 4; ++j) {
    int sp = s - 3 + j;
    if (sp >= 0) {
      const float* row = P + (b * Sn + sp) * 192;
      ka += row[c]       * kw[c * 4 + j];
      va += row[64 + c]  * vw[c * 4 + j];
      qa += row[128 + c] * qw[c * 4 + j];
    }
  }
  redk[c] = ka * ka; redq[c] = qa * qa;
  __syncthreads();
  for (int off = 32; off >= 1; off >>= 1) {
    if (c < off) { redk[c] += redk[c + off]; redq[c] += redq[c + off]; }
    __syncthreads();
  }
  float* row = KVQ + bs * 192;
  row[c]       = ka * rsqrtf(redk[0] + 1e-12f);
  row[64 + c]  = va;
  row[128 + c] = qa * rsqrtf(redq[0] + 1e-12f);
}

// ---------------------------------------------------------------------------
// Kernel 4: the 1024-step sequential scan. One block per batch sample.
// Thread t owns: W1 row t, S1 row t (VGPRs), W2^T row t (LDS, stride 65),
// S2 column t (VGPRs), b1_t / Sb1_t (scalars). k|v|q tiles arrive via
// double-buffered Tensor-Data-Mover DMA (TENSORcnt). Fused LayerNorm.
// ---------------------------------------------------------------------------
__global__ void __launch_bounds__(256)
scan_kernel(const float* __restrict__ KVQ, const float* __restrict__ G,
            const float* __restrict__ W1g, const float* __restrict__ b1g,
            const float* __restrict__ W2g, const float* __restrict__ b2g,
            const float* __restrict__ lng, const float* __restrict__ lnb,
            float* __restrict__ RN) {
  extern __shared__ float sm[];
  float* kvq   = sm;                 // 2 x 192 double buffer (TDM destination)
  float* W2T   = kvq + 384;          // 256*65 (padded: banks (t+i)%64)
  float* hbuf  = W2T + 256 * 65;     // 256
  float* dybuf = hbuf + 256;         // 64
  float* b2buf = dybuf + 64;         // 64
  float* sb2   = b2buf + 64;         // 64
  float* lngs  = sb2 + 64;           // 64
  float* lnbs  = lngs + 64;          // 64
  float* red   = lnbs + 64;          // 64
  float* gate  = red + 64;           // 8

  int t = threadIdx.x;
  int b = blockIdx.x;

  float W1r[64], S1r[64], S2c[64];
#pragma unroll
  for (int c = 0; c < 64; ++c) {
    W1r[c] = W1g[t * 64 + c];
    S1r[c] = 0.f;
    S2c[c] = 0.f;
    W2T[t * 65 + c] = W2g[c * 256 + t];   // W2 is (64,256) row-major
  }
  float b1t = b1g[t], sb1 = 0.f;
  if (t < 64) { b2buf[t] = b2g[t]; sb2[t] = 0.f; lngs[t] = lng[t]; lnbs[t] = lnb[t]; }

  // Prime the pipeline: wave 0 DMAs the first k|v|q tile into buffer 0.
  if (t < 32) tdm_load_kvq(KVQ + (size_t)b * Sn * 192, kvq);
  __syncthreads();

  for (int s = 0; s < Sn; ++s) {
    int bs = b * Sn + s;
    if (t < 6) gate[t] = G[bs * 6 + t];
    if (t == 0 && s + 1 < Sn)
      __builtin_prefetch(&G[(bs + 1) * 6], 0, 1);    // gfx1250 global_prefetch_b8
    if (t < 32) __builtin_amdgcn_s_wait_tensorcnt(0);
    __syncthreads();                                  // current kvq tile visible

    const float* kb_ = kvq + (s & 1) * 192;
    const float* vb_ = kb_ + 64;
    const float* qb_ = kb_ + 128;

    // Overlap: DMA next step's tile into the other buffer while we compute.
    if (t < 32 && s + 1 < Sn)
      tdm_load_kvq(KVQ + (size_t)(bs + 1) * 192, kvq + ((s + 1) & 1) * 192);

    // ---- forward with OLD params: h = silu(W1 k + b1) ----
    float pre = b1t;
#pragma unroll
    for (int c = 0; c < 64; ++c) pre += W1r[c] * kb_[c];
    float sg = 1.f / (1.f + __expf(-pre));
    float h  = pre * sg;
    hbuf[t] = h;
    __syncthreads();

    // ---- y = W2 h + b2 + k ;  dy = 2(y - v) ----
    if (t < 64) {
      float acc = b2buf[t] + kb_[t];
      for (int u = 0; u < 256; ++u) acc += W2T[u * 65 + t] * hbuf[u];
      dybuf[t] = 2.f * (acc - vb_[t]);
    }
    __syncthreads();

    float th0 = gate[0], th1 = gate[1];
    float et0 = gate[2], et1 = gate[3];
    float al0 = gate[4], al1 = gate[5];

    // dh_t uses OLD W2 (own row only -> race-free before own update)
    float dh = 0.f;
#pragma unroll
    for (int i = 0; i < 64; ++i) dh += W2T[t * 65 + i] * dybuf[i];
    float dpre = dh * (sg * (1.f + pre * (1.f - sg)));   // silu'

    // S = eta*S - theta*g ; W = (1-alpha)*W + S   (gate idx map 0,1,1,1)
#pragma unroll
    for (int c = 0; c < 64; ++c) {
      float g1 = dpre * kb_[c];
      S1r[c] = et0 * S1r[c] - th0 * g1;
      W1r[c] = (1.f - al0) * W1r[c] + S1r[c];
    }
    sb1 = et1 * sb1 - th1 * dpre;
    b1t = (1.f - al1) * b1t + sb1;
#pragma unroll
    for (int i = 0; i < 64; ++i) {
      float g2 = dybuf[i] * h;
      S2c[i] = et1 * S2c[i] - th1 * g2;
      W2T[t * 65 + i] = (1.f - al1) * W2T[t * 65 + i] + S2c[i];
    }
    if (t < 64) {
      sb2[t]   = et1 * sb2[t] - th1 * dybuf[t];
      b2buf[t] = (1.f - al1) * b2buf[t] + sb2[t];
    }
    __syncthreads();

    // ---- retrieve with NEW params ----
    float pre2 = b1t;
#pragma unroll
    for (int c = 0; c < 64; ++c) pre2 += W1r[c] * qb_[c];
    float sg2 = 1.f / (1.f + __expf(-pre2));
    hbuf[t] = pre2 * sg2;
    __syncthreads();

    float rv = 0.f;
    if (t < 64) {
      float acc = b2buf[t] + qb_[t];
      for (int u = 0; u < 256; ++u) acc += W2T[u * 65 + t] * hbuf[u];
      rv = acc;
    }
    // ---- fused LayerNorm over DV=64 ----
    if (t < 64) red[t] = rv;
    __syncthreads();
    for (int off = 32; off >= 1; off >>= 1) { if (t < off) red[t] += red[t + off]; __syncthreads(); }
    float mu = red[0] * (1.f / 64.f);
    __syncthreads();
    if (t < 64) { float d = rv - mu; red[t] = d * d; }
    __syncthreads();
    for (int off = 32; off >= 1; off >>= 1) { if (t < off) red[t] += red[t + off]; __syncthreads(); }
    float var = red[0] * (1.f / 64.f);
    __syncthreads();
    if (t < 64)
      RN[bs * 64 + t] = (rv - mu) * rsqrtf(var + 1e-5f) * lngs[t] + lnbs[t];
    __syncthreads();
  }
}

// ---------------------------------------------------------------------------
// Kernel 5: out = RN @ Wo^T + bo   (4096x64 @ 64x1024, WMMA)
// grid 4096 x block 128; 16384 tiles, K-loop 64/32
// ---------------------------------------------------------------------------
__global__ void __launch_bounds__(128)
outproj_kernel(const float* __restrict__ RN,
               const float* __restrict__ Wo, const float* __restrict__ bo,
               float* __restrict__ out) {
  int wave = threadIdx.x >> 5;
  int lane = threadIdx.x & 31;
  int tile = blockIdx.x * 4 + wave;          // 256 row-tiles * 64 col-tiles
  int rowt = tile >> 6, colt = tile & 63;
  int row0 = rowt * 16, n0 = colt * 16;

  v8f c = {};
#pragma unroll
  for (int k0 = 0; k0 < 64; k0 += 32) {
    v16h a = load_a_frag(RN + row0 * 64, 64, k0, lane);
    v16h b = load_b_frag(Wo, 64, n0, k0, lane);
    c = __builtin_amdgcn_wmma_f32_16x16x32_f16(false, a, false, b,
                                               (short)0, c, false, false);
  }
  int   col   = lane & 15;
  float badd  = bo[n0 + col];
  int   mbase = (lane < 16) ? 0 : 8;
#pragma unroll
  for (int r = 0; r < 8; ++r)
    out[(row0 + mbase + r) * Dn + n0 + col] = c[r] + badd;
}

// ---------------------------------------------------------------------------
extern "C" void kernel_launch(void* const* d_in, const int* in_sizes, int n_in,
                              void* d_out, int out_size, void* d_ws, size_t ws_size,
                              hipStream_t stream) {
  const float* x    = (const float*)d_in[0];
  const float* Wk   = (const float*)d_in[1];  const float* bk   = (const float*)d_in[2];
  const float* Wv   = (const float*)d_in[3];  const float* bv   = (const float*)d_in[4];
  const float* Wq   = (const float*)d_in[5];  const float* bq   = (const float*)d_in[6];
  const float* kw   = (const float*)d_in[7];  const float* kb   = (const float*)d_in[8];
  const float* vw   = (const float*)d_in[9];  const float* vb   = (const float*)d_in[10];
  const float* qw   = (const float*)d_in[11]; const float* qb   = (const float*)d_in[12];
  const float* Wlr  = (const float*)d_in[13]; const float* blr  = (const float*)d_in[14];
  const float* Wmom = (const float*)d_in[15]; const float* bmom = (const float*)d_in[16];
  const float* Wfg  = (const float*)d_in[17]; const float* bfg  = (const float*)d_in[18];
  const float* W1   = (const float*)d_in[19]; const float* b1   = (const float*)d_in[20];
  const float* W2   = (const float*)d_in[21]; const float* b2   = (const float*)d_in[22];
  const float* lng  = (const float*)d_in[23]; const float* lnb  = (const float*)d_in[24];
  const float* Wo   = (const float*)d_in[25]; const float* bo   = (const float*)d_in[26];

  float* out = (float*)d_out;
  float* ws  = (float*)d_ws;
  float* P   = ws; ws += (size_t)BSn * 192;   // raw k|v|q projections
  float* G   = ws; ws += (size_t)BSn * 6;     // theta,eta,alpha
  float* KVQ = ws; ws += (size_t)BSn * 192;   // conv'd + normalized k|v|q
  float* RN  = ws; ws += (size_t)BSn * 64;    // post-LN retrieved

  proj_kernel<<<768, 128, 0, stream>>>(x, Wk, bk, Wv, bv, Wq, bq, P);
  gates_kernel<<<BSn, 192, 0, stream>>>(x, Wlr, blr, Wmom, bmom, Wfg, bfg, G);
  conv_kernel<<<BSn, 64, 0, stream>>>(P, kw, kb, vw, vb, qw, qb, KVQ);

  size_t smem = (size_t)(384 + 256 * 65 + 256 + 64 * 6 + 8) * sizeof(float); // ~70 KB (gfx1250: 320 KB/WGP)
  (void)hipFuncSetAttribute(reinterpret_cast<const void*>(scan_kernel),
                            hipFuncAttributeMaxDynamicSharedMemorySize, (int)smem);
  scan_kernel<<<Bn, 256, smem, stream>>>(KVQ, G, W1, b1, W2, b2, lng, lnb, RN);

  outproj_kernel<<<4096, 128, 0, stream>>>(RN, Wo, bo, out);
}
